// PNA_37580963840346
// MI455X (gfx1250) — matrix operand order repaired and checked by reading, not compiled, not run on hardware
//
#include <hip/hip_runtime.h>
#include <hip/hip_bf16.h>

#define HID  50
#define HPAD 64

typedef __attribute__((ext_vector_type(16))) __bf16 v16bf;
typedef __attribute__((ext_vector_type(8)))  float  v8f;

// ---------- helpers ----------
__device__ __forceinline__ unsigned short f2bf(float x) {
  unsigned u = __float_as_uint(x);
  unsigned r = u + 0x7FFFu + ((u >> 16) & 1u);   // round-to-nearest-even
  return (unsigned short)(r >> 16);
}
// order-preserving float -> uint encoding (for atomicMin/Max on floats)
__device__ __forceinline__ unsigned encf(float f) {
  unsigned u = __float_as_uint(f);
  return ((int)u >= 0) ? (u | 0x80000000u) : ~u;
}
__device__ __forceinline__ float decf(unsigned u) {
  return __uint_as_float((u & 0x80000000u) ? (u & 0x7FFFFFFFu) : ~u);
}
#define ENC_PINF 0xFF800000u
#define ENC_NINF 0x007FFFFFu

// ---------- init all scratch ----------
__global__ void init_ws(float* deg, float* s1s, float* s1s2,
                        unsigned* s1mn, unsigned* s1mx, float* avgSum,
                        float* sum2, float* sumsq2, unsigned* mn2, unsigned* mx2,
                        float* gsum, float* gcnt, int N, int G) {
  int i = blockIdx.x * blockDim.x + threadIdx.x;
  int NT = N * HID;
  if (i < NT) { sum2[i] = 0.f; sumsq2[i] = 0.f; mn2[i] = ENC_PINF; mx2[i] = ENC_NINF; }
  if (i < N)  { deg[i] = 0.f; s1s[i] = 0.f; s1s2[i] = 0.f; s1mn[i] = ENC_PINF; s1mx[i] = ENC_NINF; }
  if (i < G * HID) gsum[i] = 0.f;
  if (i < G)  gcnt[i] = 0.f;
  if (i == 0) avgSum[0] = 0.f;
}

// ---------- layer-1 edge messages (F=1, scalar) + degree ----------
__global__ void l1_edge(const float* __restrict__ x,
                        const int* __restrict__ src, const int* __restrict__ dst,
                        const float* __restrict__ pre1_w, const float* __restrict__ pre1_b,
                        float* deg, float* s1s, float* s1s2,
                        unsigned* s1mn, unsigned* s1mx, int E) {
  int e = blockIdx.x * blockDim.x + threadIdx.x;
  if (e >= E) return;
  int d = dst[e];
  float m = pre1_w[0] * x[d] + pre1_w[1] * x[src[e]] + pre1_b[0];
  atomicAdd(&deg[d], 1.f);
  atomicAdd(&s1s[d], m);
  atomicAdd(&s1s2[d], m * m);
  unsigned ev = encf(m);
  atomicMin(&s1mn[d], ev);
  atomicMax(&s1mx[d], ev);
}

// ---------- avg_log = mean(log(deg+1)) ----------
__global__ void avg_log_sum(const float* __restrict__ deg, float* avgSum, int N) {
  __shared__ float red[256];
  int v = blockIdx.x * 256 + threadIdx.x;
  red[threadIdx.x] = (v < N) ? logf(deg[v] + 1.f) : 0.f;
  __syncthreads();
  for (int s = 128; s > 0; s >>= 1) {
    if ((int)threadIdx.x < s) red[threadIdx.x] += red[threadIdx.x + s];
    __syncthreads();
  }
  if (threadIdx.x == 0) atomicAdd(avgSum, red[0]);
}

// ---------- layer-1 node update: stats -> 13-vec -> post1 -> lin1 -> relu ----------
// Also emits h1 rows in bf16, padded to 64, with a constant-1.0 feature at
// slot 50 so layer-2's edge bias can be folded into the packed weights.
__global__ void __launch_bounds__(128) l1_node(
    const float* __restrict__ x, const float* __restrict__ deg,
    const float* __restrict__ s1s, const float* __restrict__ s1s2,
    const unsigned* __restrict__ s1mn, const unsigned* __restrict__ s1mx,
    const float* __restrict__ avgSum,
    const float* __restrict__ post1_w, const float* __restrict__ post1_b,
    const float* __restrict__ lin1_w, const float* __restrict__ lin1_b,
    float* __restrict__ h1, unsigned short* __restrict__ h1bf, int N) {
  __shared__ float ts[128 * 51];
  int v = blockIdx.x * 128 + threadIdx.x;
  if (v >= N) return;
  float dg = deg[v];
  float cnt = fmaxf(dg, 1.f);
  float mean = s1s[v] / cnt;
  float mean2 = s1s2[v] / cnt;
  float stdv = sqrtf(fmaxf(mean2 - mean * mean, 0.f) + 1e-5f);
  bool has = dg > 0.f;
  float mn = has ? decf(s1mn[v]) : 0.f;
  float mx = has ? decf(s1mx[v]) : 0.f;
  float logd = logf(cnt + 1.f);
  float avg = avgSum[0] / (float)N;
  float sa = logd / avg, sb = avg / logd;
  float vec[13];
  vec[0] = x[v];
  vec[1] = mean; vec[2] = mn; vec[3] = mx; vec[4] = stdv;
#pragma unroll
  for (int k = 0; k < 4; ++k) { vec[5 + k] = vec[1 + k] * sa; vec[9 + k] = vec[1 + k] * sb; }
  float* t = &ts[threadIdx.x * 51];
  for (int j = 0; j < HID; ++j) {
    float acc = post1_b[j];
#pragma unroll
    for (int k = 0; k < 13; ++k) acc += vec[k] * post1_w[k * HID + j];
    t[j] = acc;
  }
  for (int j = 0; j < HID; ++j) {
    float acc = lin1_b[j];
    for (int k = 0; k < HID; ++k) acc += t[k] * lin1_w[k * HID + j];
    acc = fmaxf(acc, 0.f);
    h1[(size_t)v * HID + j] = acc;
    h1bf[(size_t)v * HPAD + j] = f2bf(acc);
  }
  h1bf[(size_t)v * HPAD + HID] = 0x3F80u;          // constant-1.0 bias feature
  for (int j = HID + 1; j < HPAD; ++j) h1bf[(size_t)v * HPAD + j] = 0;
}

// ---------- pre-pack W^T (+bias row) into WMMA A-fragment layout (bf16) ----------
// fragment f = (mat*2 + kstep)*4 + rowtile; A is 16(out-ch) x 32(K):
//   lane L: M(out-ch within tile) = L&15, hi = L>>4
//   element j: K = kstep*32 + hi*16 + j
//   n (global out-ch) = rowtile*16 + M
//   value = pre2_w[mat*50 + K][n]; K==50 && mat==0 -> pre2_b[n] (bias row)
__global__ void prepack(const float* __restrict__ pre2_w, const float* __restrict__ pre2_b,
                        unsigned short* __restrict__ wpack) {
  int i = blockIdx.x * blockDim.x + threadIdx.x;
  if (i >= 16 * 512) return;
  int f = i >> 9;
  int lane = (i >> 4) & 31;
  int j = i & 15;
  int rt = f & 3;
  int ks = (f >> 2) & 1;
  int mat = f >> 3;
  int hi = lane >> 4;
  int K = ks * 32 + hi * 16 + j;
  int n = rt * 16 + (lane & 15);
  float v = 0.f;
  if (n < HID) {
    if (K < HID) v = pre2_w[(mat * HID + K) * HID + n];
    else if (K == HID && mat == 0) v = pre2_b[n];
  }
  wpack[i] = f2bf(v);
}

// ---------- layer-2 edge messages: weights-as-A, gathered-edges-as-B WMMA ----------
// C tile: lane L holds edge (L&15) and 8 consecutive out-channels (rt*16 + hi*8 + r)
// -> single dst index per lane, immediate-offset atomics.
__global__ void __launch_bounds__(256) l2_edge_wmma(
    const int* __restrict__ srcIdx, const int* __restrict__ dstIdx,
    const unsigned short* __restrict__ h1bf, const unsigned short* __restrict__ wpack,
    float* __restrict__ sum2, float* __restrict__ sumsq2,
    unsigned* __restrict__ mn2, unsigned* __restrict__ mx2, int E) {
  const int lane = threadIdx.x & 31;
  const int m = lane & 15;
  const int hi = lane >> 4;
  const int wave = (int)((blockIdx.x * blockDim.x + threadIdx.x) >> 5);
  const int nwaves = (int)((gridDim.x * blockDim.x) >> 5);

  // weight A-fragments (2 mats x 2 k-steps x 4 row-tiles), hoisted into VGPRs
  v16bf Aw[16];
#pragma unroll
  for (int f = 0; f < 16; ++f)
    Aw[f] = *reinterpret_cast<const v16bf*>(wpack + (size_t)f * 512 + lane * 16);

  const int ntiles = E >> 4;
  for (int t = wave; t < ntiles; t += nwaves) {
    const int e0 = t << 4;
    const int tn = t + nwaves;
    if (tn < ntiles) {
      __builtin_prefetch(&dstIdx[tn << 4], 0, 1);
      __builtin_prefetch(&srcIdx[tn << 4], 0, 1);
    }

    const int rd = dstIdx[e0 + m];
    const int rs = srcIdx[e0 + m];

    // B gathers: lane holds edge m's features, K = ks*32 + hi*16 + (0..15)
    union BF { uint4 q[2]; v16bf v; };
    BF bD0, bD1, bS0, bS1;
    {
      const size_t bd = (size_t)rd * HPAD + hi * 16;   // ushort units; 16B aligned
      const size_t bs = (size_t)rs * HPAD + hi * 16;
      bD0.q[0] = *(const uint4*)(h1bf + bd);      bD0.q[1] = *(const uint4*)(h1bf + bd + 8);
      bD1.q[0] = *(const uint4*)(h1bf + bd + 32); bD1.q[1] = *(const uint4*)(h1bf + bd + 40);
      bS0.q[0] = *(const uint4*)(h1bf + bs);      bS0.q[1] = *(const uint4*)(h1bf + bs + 8);
      bS1.q[0] = *(const uint4*)(h1bf + bs + 32); bS1.q[1] = *(const uint4*)(h1bf + bs + 40);
    }

    v8f c[4];
#pragma unroll
    for (int rt = 0; rt < 4; ++rt) {
      v8f acc = {0.f, 0.f, 0.f, 0.f, 0.f, 0.f, 0.f, 0.f};
      acc = __builtin_amdgcn_wmma_f32_16x16x32_bf16(false, Aw[0 * 4 + rt], false, bD0.v, (short)0, acc, false, false);
      acc = __builtin_amdgcn_wmma_f32_16x16x32_bf16(false, Aw[1 * 4 + rt], false, bD1.v, (short)0, acc, false, false);
      acc = __builtin_amdgcn_wmma_f32_16x16x32_bf16(false, Aw[2 * 4 + rt], false, bS0.v, (short)0, acc, false, false);
      acc = __builtin_amdgcn_wmma_f32_16x16x32_bf16(false, Aw[3 * 4 + rt], false, bS1.v, (short)0, acc, false, false);
      c[rt] = acc;
    }

    // writeback: channels rt*16 + hi*8 + r of edge m's dst row (bias already in C)
    const unsigned base = (unsigned)rd * (unsigned)HID + (unsigned)(hi * 8);
#pragma unroll
    for (int rt = 0; rt < 3; ++rt) {
#pragma unroll
      for (int r = 0; r < 8; ++r) {
        const float val = c[rt][r];
        const unsigned off = base + (unsigned)(rt * 16 + r);
        atomicAdd(&sum2[off], val);
        atomicAdd(&sumsq2[off], val * val);
        const unsigned ev = encf(val);
        atomicMin(&mn2[off], ev);
        atomicMax(&mx2[off], ev);
      }
    }
    if (hi == 0) {   // row-tile 3: only channels 48,49 are real
#pragma unroll
      for (int r = 0; r < 2; ++r) {
        const float val = c[3][r];
        const unsigned off = base + (unsigned)(48 + r);
        atomicAdd(&sum2[off], val);
        atomicAdd(&sumsq2[off], val * val);
        const unsigned ev = encf(val);
        atomicMin(&mn2[off], ev);
        atomicMax(&mx2[off], ev);
      }
    }
  }
}

// ---------- scalar tail for E % 16 ----------
__global__ void l2_edge_tail(const int* __restrict__ src, const int* __restrict__ dst,
                             const float* __restrict__ h1,
                             const float* __restrict__ pre2_w, const float* __restrict__ pre2_b,
                             float* sum2, float* sumsq2, unsigned* mn2, unsigned* mx2,
                             int eStart, int E) {
  int e = eStart + blockIdx.x * blockDim.x + threadIdx.x;
  if (e >= E) return;
  int rd = dst[e], rs = src[e];
  for (int j = 0; j < HID; ++j) {
    float acc = pre2_b[j];
    for (int k = 0; k < HID; ++k)
      acc += h1[(size_t)rd * HID + k] * pre2_w[k * HID + j]
           + h1[(size_t)rs * HID + k] * pre2_w[(HID + k) * HID + j];
    unsigned off = (unsigned)rd * HID + j;
    atomicAdd(&sum2[off], acc);
    atomicAdd(&sumsq2[off], acc * acc);
    unsigned ev = encf(acc);
    atomicMin(&mn2[off], ev);
    atomicMax(&mx2[off], ev);
  }
}

// ---------- layer-2 node update: stats -> (650->50) with folded scalers -> lin2 -> relu ----------
__global__ void __launch_bounds__(64) l2_node(
    const float* __restrict__ h1, const float* __restrict__ deg,
    const float* __restrict__ sum2, const float* __restrict__ sumsq2,
    const unsigned* __restrict__ mn2, const unsigned* __restrict__ mx2,
    const float* __restrict__ avgSum,
    const float* __restrict__ post2_w, const float* __restrict__ post2_b,
    const float* __restrict__ lin2_w, const float* __restrict__ lin2_b,
    float* __restrict__ h2, int N) {
  int v = blockIdx.x;
  if (v >= N) return;
  int tid = threadIdx.x;
  __shared__ float agg[4 * HID];
  __shared__ float hv[HID];
  __shared__ float t[HID];
  float dg = deg[v];
  float cnt = fmaxf(dg, 1.f);
  bool has = dg > 0.f;
  float logd = logf(cnt + 1.f);
  float avg = avgSum[0] / (float)N;
  float sa = logd / avg, sb = avg / logd;
  if (tid < HID) {
    unsigned off = (unsigned)v * HID + tid;
    hv[tid] = h1[off];
    float mean = sum2[off] / cnt;
    float m2 = sumsq2[off] / cnt;
    float stdv = sqrtf(fmaxf(m2 - mean * mean, 0.f) + 1e-5f);
    agg[tid] = mean;
    agg[HID + tid] = has ? decf(mn2[off]) : 0.f;
    agg[2 * HID + tid] = has ? decf(mx2[off]) : 0.f;
    agg[3 * HID + tid] = stdv;
  }
  __syncthreads();
  if (tid < HID) {
    float acc = post2_b[tid];
    for (int k = 0; k < HID; ++k) acc += hv[k] * post2_w[k * HID + tid];
    for (int k = 0; k < 4 * HID; ++k) {
      float w = post2_w[(HID + k) * HID + tid]
              + sa * post2_w[(5 * HID + k) * HID + tid]
              + sb * post2_w[(9 * HID + k) * HID + tid];
      acc += agg[k] * w;
    }
    t[tid] = acc;
  }
  __syncthreads();
  if (tid < HID) {
    float acc = lin2_b[tid];
    for (int k = 0; k < HID; ++k) acc += t[k] * lin2_w[k * HID + tid];
    h2[(size_t)v * HID + tid] = fmaxf(acc, 0.f);
  }
}

// ---------- global mean pool (batch is sorted: wave-uniform fast path) ----------
__global__ void pool(const float* __restrict__ h2, const int* __restrict__ batch,
                     float* gsum, float* gcnt, int N) {
  int v = blockIdx.x * blockDim.x + threadIdx.x;
  int lane = threadIdx.x & 31;
  bool valid = v < N;
  int vc = valid ? v : (N - 1);
  int g = batch[vc];
  int g0 = __shfl(g, 0);
  int g31 = __shfl(g, 31);
  if (__all(valid) && g0 == g31) {
    // whole wave belongs to one graph: shuffle-reduce, one atomic per channel
    if (lane == 0) atomicAdd(&gcnt[g0], 32.f);
    for (int j = 0; j < HID; ++j) {
      float r = h2[(size_t)v * HID + j];
      r += __shfl_down(r, 16);
      r += __shfl_down(r, 8);
      r += __shfl_down(r, 4);
      r += __shfl_down(r, 2);
      r += __shfl_down(r, 1);
      if (lane == 0) atomicAdd(&gsum[(unsigned)g0 * HID + j], r);
    }
  } else {
    if (!valid) return;
    atomicAdd(&gcnt[g], 1.f);
    for (int j = 0; j < HID; ++j)
      atomicAdd(&gsum[(unsigned)g * HID + j], h2[(size_t)v * HID + j]);
  }
}

// ---------- head: linear + sigmoid ----------
__global__ void head(const float* __restrict__ gsum, const float* __restrict__ gcnt,
                     const float* __restrict__ out_w, const float* __restrict__ out_b,
                     float* __restrict__ out, int G) {
  int g = blockIdx.x * blockDim.x + threadIdx.x;
  if (g >= G) return;
  float c = fmaxf(gcnt[g], 1.f);
  float acc = out_b[0];
  for (int j = 0; j < HID; ++j) acc += (gsum[g * HID + j] / c) * out_w[j];
  out[g] = 1.f / (1.f + expf(-acc));
}

extern "C" void kernel_launch(void* const* d_in, const int* in_sizes, int n_in,
                              void* d_out, int out_size, void* d_ws, size_t ws_size,
                              hipStream_t stream) {
  const float* x       = (const float*)d_in[0];
  const int*   ei      = (const int*)d_in[1];
  const int*   batch   = (const int*)d_in[2];
  const float* pre1_w  = (const float*)d_in[3];
  const float* pre1_b  = (const float*)d_in[4];
  const float* post1_w = (const float*)d_in[5];
  const float* post1_b = (const float*)d_in[6];
  const float* lin1_w  = (const float*)d_in[7];
  const float* lin1_b  = (const float*)d_in[8];
  const float* pre2_w  = (const float*)d_in[9];
  const float* pre2_b  = (const float*)d_in[10];
  const float* post2_w = (const float*)d_in[11];
  const float* post2_b = (const float*)d_in[12];
  const float* lin2_w  = (const float*)d_in[13];
  const float* lin2_b  = (const float*)d_in[14];
  const float* out_w   = (const float*)d_in[15];
  const float* out_b   = (const float*)d_in[16];

  const int N = in_sizes[0];
  const int E = in_sizes[1] / 2;
  const int G = out_size;
  const int* srcIdx = ei;
  const int* dstIdx = ei + E;

  char* base = (char*)d_ws;
  size_t off = 0;
  auto carve = [&](size_t bytes) -> char* {
    char* p = base + off;
    off = (off + bytes + 255) & ~(size_t)255;
    return p;
  };
  float*          deg    = (float*)carve((size_t)N * 4);
  float*          s1s    = (float*)carve((size_t)N * 4);
  float*          s1s2   = (float*)carve((size_t)N * 4);
  unsigned*       s1mn   = (unsigned*)carve((size_t)N * 4);
  unsigned*       s1mx   = (unsigned*)carve((size_t)N * 4);
  float*          avgSum = (float*)carve(256);
  float*          h1     = (float*)carve((size_t)N * HID * 4);
  unsigned short* h1bf   = (unsigned short*)carve((size_t)N * HPAD * 2);
  unsigned short* wpack  = (unsigned short*)carve(16 * 512 * 2);
  float*          sum2   = (float*)carve((size_t)N * HID * 4);
  float*          sumsq2 = (float*)carve((size_t)N * HID * 4);
  unsigned*       mn2    = (unsigned*)carve((size_t)N * HID * 4);
  unsigned*       mx2    = (unsigned*)carve((size_t)N * HID * 4);
  float*          h2     = (float*)carve((size_t)N * HID * 4);
  float*          gsum   = (float*)carve((size_t)G * HID * 4);
  float*          gcnt   = (float*)carve((size_t)G * 4);
  (void)ws_size; (void)n_in;

  int NT = N * HID;
  init_ws<<<(NT + 255) / 256, 256, 0, stream>>>(deg, s1s, s1s2, s1mn, s1mx, avgSum,
                                                sum2, sumsq2, mn2, mx2, gsum, gcnt, N, G);
  l1_edge<<<(E + 255) / 256, 256, 0, stream>>>(x, srcIdx, dstIdx, pre1_w, pre1_b,
                                               deg, s1s, s1s2, s1mn, s1mx, E);
  avg_log_sum<<<(N + 255) / 256, 256, 0, stream>>>(deg, avgSum, N);
  l1_node<<<(N + 127) / 128, 128, 0, stream>>>(x, deg, s1s, s1s2, s1mn, s1mx, avgSum,
                                               post1_w, post1_b, lin1_w, lin1_b, h1, h1bf, N);
  prepack<<<32, 256, 0, stream>>>(pre2_w, pre2_b, wpack);

  int tiles = E >> 4;
  if (tiles > 0)
    l2_edge_wmma<<<1024, 256, 0, stream>>>(srcIdx, dstIdx, h1bf, wpack,
                                           sum2, sumsq2, mn2, mx2, E);
  int rem = E - (tiles << 4);
  if (rem > 0)
    l2_edge_tail<<<(rem + 127) / 128, 128, 0, stream>>>(srcIdx, dstIdx, h1, pre2_w, pre2_b,
                                                        sum2, sumsq2, mn2, mx2, tiles << 4, E);

  l2_node<<<N, 64, 0, stream>>>(h1, deg, sum2, sumsq2, mn2, mx2, avgSum,
                                post2_w, post2_b, lin2_w, lin2_b, h2, N);
  pool<<<(N + 255) / 256, 256, 0, stream>>>(h2, batch, gsum, gcnt, N);
  head<<<1, 256, 0, stream>>>(gsum, gcnt, out_w, out_b, (float*)d_out, G);
}
